// LSTM_77841987273040
// MI455X (gfx1250) — matrix operand order repaired and checked by reading, not compiled, not run on hardware
//
#include <hip/hip_runtime.h>

// ---------------------------------------------------------------------------
// CDNA5 (gfx1250) LSTM forward: bf16 WMMA GEMMs with f32 accumulation.
// H=1024, L=2, B=64, S=512, LBL=5.
// ---------------------------------------------------------------------------

typedef __attribute__((ext_vector_type(16))) __bf16 v16bf;
typedef __attribute__((ext_vector_type(8)))  float  v8f;

#define HID   1024
#define G4    4096
#define BATCH 64
#define SEQ   512
#define SB    (SEQ * BATCH)      // 32768 rows of the xg GEMM

__device__ inline v8f vzero8() {
  v8f z;
#pragma unroll
  for (int r = 0; r < 8; r++) z[r] = 0.0f;
  return z;
}

__device__ inline v8f wmma_bf16(v16bf a, v16bf b, v8f c) {
  // (neg_a, A, neg_b, B, c_mod, C, reuse_a, reuse_b)
  return __builtin_amdgcn_wmma_f32_16x16x32_bf16(false, a, false, b,
                                                 (short)0, c, false, false);
}

// A fragment: 16x32 bf16, row-major tile in LDS, row stride lda elements.
// Per ISA 7.12.2: lane m (0..15) VGPR0-3 hold K={0..7} (2/VGPR), VGPR4-7 K={16..23};
// lane m+16 holds K={8..15} and K={24..31}.
__device__ inline v16bf load_frag_a(const __bf16* base, int lda) {
  const int lane = threadIdx.x & 31;
  const int m    = lane & 15;
  const int half = lane >> 4;
  union { v16bf v; unsigned u[8]; } r;
  const __bf16* rowp = base + m * lda;
  const int kb0 = half * 8;
  const int kb1 = 16 + half * 8;
#pragma unroll
  for (int i = 0; i < 4; i++) r.u[i]     = *(const unsigned*)(rowp + kb0 + 2 * i);
#pragma unroll
  for (int i = 0; i < 4; i++) r.u[4 + i] = *(const unsigned*)(rowp + kb1 + 2 * i);
  return r.v;
}

// B fragment: 32x16 bf16 (K x N). LDS tile stored transposed [n][k], row stride ldb.
// Lane n (0..15): K={0..7},{16..23}; lane n+16: K={8..15},{24..31} (2/VGPR).
__device__ inline v16bf load_frag_b(const __bf16* base, int ldb) {
  const int lane = threadIdx.x & 31;
  const int n    = lane & 15;
  const int half = lane >> 4;
  union { v16bf v; unsigned u[8]; } r;
  const __bf16* rowp = base + n * ldb;
  const int kb0 = half * 8;
  const int kb1 = 16 + half * 8;
#pragma unroll
  for (int i = 0; i < 4; i++) r.u[i]     = *(const unsigned*)(rowp + kb0 + 2 * i);
#pragma unroll
  for (int i = 0; i < 4; i++) r.u[4 + i] = *(const unsigned*)(rowp + kb1 + 2 * i);
  return r.v;
}

__device__ inline float sigf(float x) { return 1.0f / (1.0f + __expf(-x)); }

// ---------------------------------------------------------------------------
// xg GEMM: C[M=32768, N=4096] = A[M,1024](bf16) * Bw[1024,4096](bf16) + bias
// 128x128x32 tiles, 8 waves, each wave: 2x4 grid of 16x16 accumulators.
// ---------------------------------------------------------------------------
__global__ __launch_bounds__(256) void gemm_xg(const __bf16* __restrict__ A,
                                               const __bf16* __restrict__ Bw,
                                               const float*  __restrict__ bias,
                                               float* __restrict__ C) {
  __shared__ __bf16 As[128 * 40];
  __shared__ __bf16 Bs[128 * 40];
  const int tid  = threadIdx.x;
  const int lane = tid & 31;
  const int wid  = tid >> 5;
  const int half = lane >> 4;
  const int l16  = lane & 15;
  const int M0   = blockIdx.y * 128;
  const int N0   = blockIdx.x * 128;
  const int wm   = wid >> 1;   // 0..3 -> 32 rows each
  const int wn   = wid & 1;    // 0..1 -> 64 cols each

  v8f acc[2][4];
#pragma unroll
  for (int i = 0; i < 2; i++)
#pragma unroll
    for (int j = 0; j < 4; j++) acc[i][j] = vzero8();

  for (int k0 = 0; k0 < HID; k0 += 32) {
    // Stage A: 128x32 bf16, 16B chunks, 2 per thread.
#pragma unroll
    for (int t = 0; t < 2; t++) {
      int ch  = tid + t * 256;
      int row = ch >> 2;
      int kc  = (ch & 3) * 8;
      *(uint4*)&As[row * 40 + kc] =
          *(const uint4*)(A + (size_t)(M0 + row) * HID + k0 + kc);
    }
    // Stage B (transpose to [n][k]): 32x128 bf16.
#pragma unroll
    for (int t = 0; t < 2; t++) {
      int ch = tid + t * 256;
      int k  = ch >> 4;
      int nc = (ch & 15) * 8;
      union { uint4 q; __bf16 h[8]; } u;
      u.q = *(const uint4*)(Bw + (size_t)(k0 + k) * G4 + N0 + nc);
#pragma unroll
      for (int j = 0; j < 8; j++) Bs[(nc + j) * 40 + k] = u.h[j];
    }
    __syncthreads();

    v16bf af[2];
#pragma unroll
    for (int i = 0; i < 2; i++) af[i] = load_frag_a(&As[(wm * 32 + i * 16) * 40], 40);
    v16bf bfv[4];
#pragma unroll
    for (int j = 0; j < 4; j++) bfv[j] = load_frag_b(&Bs[(wn * 64 + j * 16) * 40], 40);
#pragma unroll
    for (int i = 0; i < 2; i++)
#pragma unroll
      for (int j = 0; j < 4; j++) acc[i][j] = wmma_bf16(af[i], bfv[j], acc[i][j]);
    __syncthreads();
  }

  // Epilogue: D layout -> lanes 0-15 (M=r), lanes 16-31 (M=8+r), N=lane%16.
#pragma unroll
  for (int i = 0; i < 2; i++) {
    int rbase = M0 + wm * 32 + i * 16 + half * 8;
#pragma unroll
    for (int j = 0; j < 4; j++) {
      int col  = N0 + wn * 64 + j * 16 + l16;
      float bv = bias[col];
#pragma unroll
      for (int r = 0; r < 8; r++)
        C[(size_t)(rbase + r) * G4 + col] = acc[i][j][r] + bv;
    }
  }
}

// ---------------------------------------------------------------------------
// One LSTM timestep. 16 WGs x 256 threads. WG g owns hidden cols jc..jc+63 and
// computes the gate-interleaved 64x256 tile of feat = xg + h@Wh + bh, then the
// gate nonlinearity, c/h update, and (last layer) online max-pool.
// ---------------------------------------------------------------------------
__global__ __launch_bounds__(256) void lstm_step(const __bf16* __restrict__ hprev, // [64,1024]
                                                 const __bf16* __restrict__ Whb,   // [1024,4096]
                                                 const float*  __restrict__ xg,    // [64,4096] at step s
                                                 const float*  __restrict__ bh,    // [4096]
                                                 float*  __restrict__ cst,         // [64,1024]
                                                 __bf16* __restrict__ hout,        // [64,1024]
                                                 float*  __restrict__ pooled) {    // [64,1024] or null
  __shared__ float smem[BATCH * 256];       // feat buffer; staging aliased below
  __bf16* As = (__bf16*)smem;               // 64 x 40 (2560 elems)
  __bf16* Bs = ((__bf16*)smem) + 64 * 40;   // 256 x 40

  const int tid  = threadIdx.x;
  const int lane = tid & 31;
  const int wid  = tid >> 5;
  const int half = lane >> 4;
  const int l16  = lane & 15;
  const int jc   = blockIdx.x * 64;
  const int mt   = wid >> 1;                // m tile 0..3 (16 rows each)
  const int nb   = (wid & 1) * 8;           // n tile base (of 16 tiles)

  v8f acc[8];
#pragma unroll
  for (int j = 0; j < 8; j++) acc[j] = vzero8();

  for (int k0 = 0; k0 < HID; k0 += 32) {
    // Stage A: 64x32 from hprev (1 chunk/thread).
    {
      int row = tid >> 2;
      int kc  = (tid & 3) * 8;
      *(uint4*)&As[row * 40 + kc] =
          *(const uint4*)(hprev + (size_t)row * HID + k0 + kc);
    }
    // Stage B: 32 k-rows x 256 gate-interleaved cols, transposed to [n][k].
#pragma unroll
    for (int t = 0; t < 4; t++) {
      int ch = tid + t * 256;
      int k  = ch >> 5;
      int nc = (ch & 31) * 8;
      int gg = nc >> 6;
      int cc = nc & 63;
      union { uint4 q; __bf16 h[8]; } u;
      u.q = *(const uint4*)(Whb + (size_t)(k0 + k) * G4 + gg * HID + jc + cc);
#pragma unroll
      for (int j = 0; j < 8; j++) Bs[(nc + j) * 40 + k] = u.h[j];
    }
    __syncthreads();

    v16bf af = load_frag_a(&As[mt * 16 * 40], 40);
#pragma unroll
    for (int j = 0; j < 8; j++) {
      v16bf bfv = load_frag_b(&Bs[(nb + j) * 16 * 40], 40);
      acc[j] = wmma_bf16(af, bfv, acc[j]);
    }
    __syncthreads();
  }

  // feat = acc + xg + bh  ->  smem[b][ncol], ncol = gate*64 + c.
#pragma unroll
  for (int j = 0; j < 8; j++) {
    int ncol = (nb + j) * 16 + l16;
    int gg   = ncol >> 6;
    int cc   = ncol & 63;
    int gcol = gg * HID + jc + cc;
    float bhv = bh[gcol];
#pragma unroll
    for (int r = 0; r < 8; r++) {
      int b = mt * 16 + half * 8 + r;
      smem[b * 256 + ncol] = acc[j][r] + xg[(size_t)b * G4 + gcol] + bhv;
    }
  }
  __syncthreads();

  // Gates + state update: 64 rows x 64 hidden cols, 16 (b,j) pairs per thread.
#pragma unroll
  for (int it = 0; it < 16; it++) {
    int idx = it * 256 + tid;
    int b   = idx >> 6;
    int j   = idx & 63;
    float fi = smem[b * 256 + j];
    float ff = smem[b * 256 + 64 + j];
    float fg = smem[b * 256 + 128 + j];
    float fo = smem[b * 256 + 192 + j];
    float ig = sigf(fi);
    float fgate = sigf(ff);
    float og = sigf(fo);
    float gt = tanhf(fg);
    size_t ci = (size_t)b * HID + jc + j;
    float cn = fgate * cst[ci] + ig * gt;
    float h  = tanhf(cn) * og;
    cst[ci]  = cn;
    hout[ci] = (__bf16)h;
    if (pooled) pooled[ci] = fmaxf(pooled[ci], h);
  }
}

// ---------------------------------------------------------------------------
// Utility kernels
// ---------------------------------------------------------------------------
__global__ void cvt_f32_bf16(const float* __restrict__ src, __bf16* __restrict__ dst, long n) {
  for (long i = (long)blockIdx.x * blockDim.x + threadIdx.x; i < n;
       i += (long)gridDim.x * blockDim.x)
    dst[i] = (__bf16)src[i];
}

// inputs [B,S,H] f32 -> time-major [S,B,H] bf16
__global__ void cvt_input_tm(const float* __restrict__ in, __bf16* __restrict__ out, long n) {
  for (long i = (long)blockIdx.x * blockDim.x + threadIdx.x; i < n;
       i += (long)gridDim.x * blockDim.x) {
    long s = i / ((long)BATCH * HID);
    long r = i % ((long)BATCH * HID);
    long b = r / HID;
    long h = r % HID;
    out[i] = (__bf16)in[(b * SEQ + s) * HID + h];
  }
}

__global__ void fill_f32(float* p, float v, long n) {
  for (long i = (long)blockIdx.x * blockDim.x + threadIdx.x; i < n;
       i += (long)gridDim.x * blockDim.x)
    p[i] = v;
}

__global__ void fill_bf16(__bf16* p, float v, long n) {
  for (long i = (long)blockIdx.x * blockDim.x + threadIdx.x; i < n;
       i += (long)gridDim.x * blockDim.x)
    p[i] = (__bf16)v;
}

// out[b, o] = bp[o] + sum_j pooled[b,j] * Wp[j,o];  64x5, K=1024.
__global__ void classify(const float* __restrict__ pooled, const float* __restrict__ Wp,
                         const float* __restrict__ bp, float* __restrict__ out) {
  int t = threadIdx.x;
  if (t >= BATCH * 5) return;
  int b = t / 5, o = t % 5;
  float acc = bp[o];
  for (int j = 0; j < HID; j++) acc += pooled[b * HID + j] * Wp[j * 5 + o];
  out[b * 5 + o] = acc;
}

// ---------------------------------------------------------------------------
// Host launcher
// ---------------------------------------------------------------------------
extern "C" void kernel_launch(void* const* d_in, const int* in_sizes, int n_in,
                              void* d_out, int out_size, void* d_ws, size_t ws_size,
                              hipStream_t stream) {
  (void)in_sizes; (void)n_in; (void)out_size; (void)ws_size;
  const float* inputs = (const float*)d_in[0];
  const float* Wx     = (const float*)d_in[1];
  const float* bx     = (const float*)d_in[2];
  const float* Wh     = (const float*)d_in[3];
  const float* bh     = (const float*)d_in[4];
  const float* Wp     = (const float*)d_in[5];
  const float* bp     = (const float*)d_in[6];

  char* ws = (char*)d_ws;
  size_t off = 0;
  auto carve = [&](size_t bytes) -> void* {
    off = (off + 255) & ~(size_t)255;
    void* p = ws + off;
    off += bytes;
    return p;
  };
  const size_t WEL = 2UL * HID * G4;                    // elems per weight tensor
  __bf16* WxB   = (__bf16*)carve(WEL * 2);              // 16 MB
  __bf16* WhB   = (__bf16*)carve(WEL * 2);              // 16 MB
  float*  xgbuf = (float*)carve((size_t)SB * G4 * 4);   // 512 MB
  __bf16* bufA  = (__bf16*)carve((size_t)SB * HID * 2); // 64 MB
  __bf16* bufB  = (__bf16*)carve((size_t)SB * HID * 2); // 64 MB
  float*  cst   = (float*)carve((size_t)BATCH * HID * 4);
  float*  pool  = (float*)carve((size_t)BATCH * HID * 4);
  __bf16* h0    = (__bf16*)carve((size_t)BATCH * HID * 2);

  cvt_f32_bf16<<<4096, 256, 0, stream>>>(Wx, WxB, (long)WEL);
  cvt_f32_bf16<<<4096, 256, 0, stream>>>(Wh, WhB, (long)WEL);
  cvt_input_tm<<<8192, 256, 0, stream>>>(inputs, bufA, (long)SB * HID);
  fill_bf16<<<256, 256, 0, stream>>>(h0, 0.0f, (long)BATCH * HID);
  fill_f32<<<256, 256, 0, stream>>>(pool, -3.402823466e38f, (long)BATCH * HID);

  for (int l = 0; l < 2; l++) {
    const __bf16* xin  = (l == 0) ? bufA : bufB;
    __bf16*       xout = (l == 0) ? bufB : bufA;
    gemm_xg<<<dim3(G4 / 128, SB / 128), 256, 0, stream>>>(
        xin, WxB + (size_t)l * HID * G4, bx + (size_t)l * G4, xgbuf);
    fill_f32<<<256, 256, 0, stream>>>(cst, 0.0f, (long)BATCH * HID);
    for (int s = 0; s < SEQ; s++) {
      const __bf16* hp = s ? (xout + (size_t)(s - 1) * BATCH * HID) : h0;
      lstm_step<<<HID / 64, 256, 0, stream>>>(
          hp, WhB + (size_t)l * HID * G4, xgbuf + (size_t)s * BATCH * G4,
          bh + (size_t)l * G4, cst, xout + (size_t)s * BATCH * HID,
          (l == 1) ? pool : nullptr);
    }
  }
  classify<<<1, 320, 0, stream>>>(pool, Wp, bp, (float*)d_out);
}
